// IndividualRNNAutoEncoder_24953759989784
// MI455X (gfx1250) — matrix operand order repaired and checked by reading, not compiled, not run on hardware
//
#include <hip/hip_runtime.h>
#include <hip/hip_bf16.h>

typedef __attribute__((ext_vector_type(16))) _Float16 v16h;
typedef __attribute__((ext_vector_type(8)))  _Float16 v8h;
typedef __attribute__((ext_vector_type(8)))  float    v8f;

union V16 { v16h v; v8h h[2]; };

#define RNN_H 512
#define RNN_T 512
#define RNN_B 64
#define RNN_E 64
#define RNN_V 4

#if __has_builtin(__builtin_amdgcn_tanhf)
#define TANHF(x) __builtin_amdgcn_tanhf(x)   // gfx1250 v_tanh_f32
#else
#define TANHF(x) tanhf(x)
#endif

// ---- d_ws layout (offsets in _Float16 elements unless noted) ----
#define OFF_EWIH0 0                       // enc w_ih layer0 [512,64]
#define OFF_EWIH1 32768                   // enc w_ih layers 1..2 [2,512,512]
#define OFF_EWHH  557056                  // enc w_hh [3,512,512]
#define OFF_DWIH  1343488                 // dec w_ih [3,512,512]
#define OFF_DWHH  2129920                 // dec w_hh [3,512,512]
#define WEIGHT_HALFS 2916352
#define BIAS_BYTE_OFF (WEIGHT_HALFS * 2)          // 3072 floats (enc 1536 + dec 1536)
#define XS_BYTE_OFF   (BIAS_BYTE_OFF + 3072 * 4)  // xs16 [T,B,E] f16

// ---- LDS layout (half offsets) ----
#define L_H0 0
#define L_H1 32768
#define L_H2 65536
#define L_XB 98304
#define LDS_BYTES ((98304 + 4096) * 2)    // 204,800 B < 320 KB WGP LDS

// ---------------- prep kernels ----------------
__global__ void rnnae_cvt_f16(const float* __restrict__ s, _Float16* __restrict__ d, int n) {
  int i = blockIdx.x * blockDim.x + threadIdx.x;
  if (i < n) d[i] = (_Float16)s[i];
}

__global__ void rnnae_bias_sum(const float* eb_ih, const float* eb_hh,
                               const float* db_ih, const float* db_hh,
                               float* be, float* bd) {
  int i = blockIdx.x * blockDim.x + threadIdx.x;
  if (i < 1536) be[i] = eb_ih[i] + eb_hh[i];
  else if (i < 3072) { int j = i - 1536; bd[j] = db_ih[j] + db_hh[j]; }
}

__global__ void rnnae_embed(const float* __restrict__ x, const float* __restrict__ emb,
                            _Float16* __restrict__ xs) {
  int i = blockIdx.x * blockDim.x + threadIdx.x;        // i = t*B + b
  if (i >= RNN_T * RNN_B) return;
  int t = i / RNN_B, b = i % RNN_B;
  const float* xp = x + ((size_t)b * RNN_T + t) * RNN_V;
  int best = 0; float bv = xp[0];
#pragma unroll
  for (int v = 1; v < RNN_V; ++v) { float f = xp[v]; if (f > bv) { bv = f; best = v; } }
  const float* e = emb + best * RNN_E;
  _Float16* dst = xs + (size_t)i * RNN_E;
#pragma unroll 8
  for (int k = 0; k < RNN_E; ++k) dst[k] = (_Float16)e[k];
}

// ---------------- GEMM core (WMMA f16 -> f32) ----------------
// C[64,512] slice: wave owns cols [n0, n0+16), 4 row tiles. A from LDS, W from L2.
// All 4 A tiles are loaded into distinct registers first so the 4 WMMAs can issue
// back-to-back without per-WMMA DS waits.
template<int KT, int LDA>
__device__ __forceinline__ void gemm_acc(v8f c[4], const _Float16* __restrict__ W,
                                         const _Float16* __restrict__ A, int lane, int n0) {
  const int K  = KT * 32;
  const int nl = lane & 15;
  const int hi = lane >> 4;
#pragma unroll 4
  for (int kc = 0; kc < KT; ++kc) {
    // B tile 32x16 (K x N): lane = N (+K-half select), 16 contiguous k halves
    const _Float16* wp = W + (size_t)(n0 + nl) * K + kc * 32 + hi * 16;
    V16 b; b.h[0] = *(const v8h*)wp; b.h[1] = *(const v8h*)(wp + 8);
    V16 a[4];
#pragma unroll
    for (int mr = 0; mr < 4; ++mr) {
      // A tile 16x32: lane row = M, 16 contiguous k halves at k = hi*16+[0,16)
      const _Float16* ap = A + (mr * 16 + nl) * LDA + kc * 32 + hi * 16;
      a[mr].h[0] = *(const v8h*)ap;
      a[mr].h[1] = *(const v8h*)(ap + 8);
    }
#pragma unroll
    for (int mr = 0; mr < 4; ++mr)
      c[mr] = __builtin_amdgcn_wmma_f32_16x16x32_f16(false, a[mr].v, false, b.v,
                                                     (short)0, c[mr], false, false);
  }
}

// bias + tanh + f16 store of C tiles into LDS hidden buffer [64,512]
__device__ __forceinline__ void store_h(const v8f c[4], _Float16* __restrict__ H,
                                        const float* __restrict__ bsum, int lane, int n0) {
  const int nl = lane & 15;
  const int hi = lane >> 4;
  float bias = bsum[n0 + nl];
#pragma unroll
  for (int mr = 0; mr < 4; ++mr)
#pragma unroll
    for (int i = 0; i < 8; ++i) {
      int m = mr * 16 + hi * 8 + i;
      H[m * RNN_H + n0 + nl] = (_Float16)TANHF(c[mr][i] + bias);
    }
}

// ---------------- persistent RNN autoencoder kernel ----------------
__global__ __launch_bounds__(1024) void rnnae_persistent(
    const _Float16* __restrict__ W, const float* __restrict__ bse,
    const float* __restrict__ bsd, const _Float16* __restrict__ xs16,
    const float* __restrict__ proj, float* __restrict__ out) {
  extern __shared__ _Float16 sm[];
  const int tid  = threadIdx.x;
  const int lane = tid & 31;
  const int n0   = (tid >> 5) * 16;          // wave's 16-column slice

  // zero hidden states (3 x [64,512] f16)
  {
    unsigned* uz = (unsigned*)sm;
    for (int i = tid; i < 49152; i += 1024) uz[i] = 0u;
  }
  __syncthreads();

  // ================= ENCODER =================
  for (int t = 0; t < RNN_T; ++t) {
    // async-stage x[t] (64x64 f16 = 8KB) into LDS via CDNA5 async load path
    {
      const _Float16* src = xs16 + (size_t)t * (RNN_B * RNN_E) + tid * 4;
      unsigned dst = (unsigned)(size_t)(const void*)(sm + L_XB + tid * 4);
      asm volatile("global_load_async_to_lds_b64 %0, %1, off"
                   :: "v"(dst), "v"(src) : "memory");
      asm volatile("s_wait_asynccnt 0" ::: "memory");
      __syncthreads();
    }
    for (int l = 0; l < 3; ++l) {
      _Float16* hb = sm + l * 32768;
      v8f c[4] = {};
      if (l == 0) {
        gemm_acc<2, RNN_E>(c, W + OFF_EWIH0, sm + L_XB, lane, n0);
      } else {
        gemm_acc<16, RNN_H>(c, W + OFF_EWIH1 + (size_t)(l - 1) * 262144,
                            sm + (l - 1) * 32768, lane, n0);
      }
      gemm_acc<16, RNN_H>(c, W + OFF_EWHH + (size_t)l * 262144, hb, lane, n0);
      __syncthreads();                       // all reads of hb done
      store_h(c, hb, bse + l * RNN_H, lane, n0);
      __syncthreads();                       // hb visible to next layer
    }
  }

  // flip layer order of final hidden states: dec_h0[l] = enc_h[2-l]
  {
    unsigned* u0 = (unsigned*)(sm + L_H0);
    unsigned* u2 = (unsigned*)(sm + L_H2);
    for (int i = tid; i < 16384; i += 1024) {
      unsigned a = u0[i], b = u2[i];
      u0[i] = b; u2[i] = a;
    }
  }
  __syncthreads();

  // ================= DECODER =================
  for (int t = 0; t < RNN_T; ++t) {
    for (int l = 0; l < 3; ++l) {
      _Float16* hb = sm + l * 32768;
      // layer-0 input = previous step's top-layer output (old hb2); x0 = zeros at t==0
      const _Float16* Ain = (l == 0) ? (sm + L_H2) : (sm + (l - 1) * 32768);
      v8f c[4] = {};
      if (l > 0 || t > 0)
        gemm_acc<16, RNN_H>(c, W + OFF_DWIH + (size_t)l * 262144, Ain, lane, n0);
      gemm_acc<16, RNN_H>(c, W + OFF_DWHH + (size_t)l * 262144, hb, lane, n0);
      __syncthreads();
      store_h(c, hb, bsd + l * RNN_H, lane, n0);
      __syncthreads();
    }
    // projection [64,512]x[512,4] + log_softmax over V=4, all 1024 threads:
    // tid = b*16 + v*4 + q, q = K-quarter; reduce q via shfl_xor(1,2), v via (4,8)
    {
      int q = tid & 3, v = (tid >> 2) & 3, b = tid >> 4;
      const _Float16* h = sm + L_H2 + b * RNN_H + q * 128;
      const float* pw = proj + v * RNN_H + q * 128;
      float acc = 0.f;
#pragma unroll 8
      for (int k = 0; k < 128; ++k) acc += (float)h[k] * pw[k];
      acc += __shfl_xor(acc, 1);
      acc += __shfl_xor(acc, 2);          // full dot for (b,v)
      float m = acc;
      m = fmaxf(m, __shfl_xor(m, 4));
      m = fmaxf(m, __shfl_xor(m, 8));
      float e = __expf(acc - m);
      float s = e;
      s += __shfl_xor(s, 4);
      s += __shfl_xor(s, 8);
      if (q == 0)
        out[((size_t)b * RNN_T + t) * RNN_V + v] = (acc - m) - __logf(s);
    }
    __syncthreads();
  }
}

// ---------------- launch ----------------
extern "C" void kernel_launch(void* const* d_in, const int* in_sizes, int n_in,
                              void* d_out, int out_size, void* d_ws, size_t ws_size,
                              hipStream_t stream) {
  const float* x        = (const float*)d_in[0];
  const float* emb      = (const float*)d_in[1];
  const float* ewih0    = (const float*)d_in[2];
  const float* ewihr    = (const float*)d_in[3];
  const float* ewhh     = (const float*)d_in[4];
  const float* ebih     = (const float*)d_in[5];
  const float* ebhh     = (const float*)d_in[6];
  const float* dwih     = (const float*)d_in[7];
  const float* dwhh     = (const float*)d_in[8];
  const float* dbih     = (const float*)d_in[9];
  const float* dbhh     = (const float*)d_in[10];
  const float* proj     = (const float*)d_in[11];

  char* ws = (char*)d_ws;
  _Float16* Wh   = (_Float16*)ws;
  float*    bse  = (float*)(ws + BIAS_BYTE_OFF);
  float*    bsd  = bse + 1536;
  _Float16* xs16 = (_Float16*)(ws + XS_BYTE_OFF);

  // f32 -> f16 weight conversion (L2-resident thereafter)
  rnnae_cvt_f16<<<(32768  + 255) / 256, 256, 0, stream>>>(ewih0, Wh + OFF_EWIH0, 32768);
  rnnae_cvt_f16<<<(524288 + 255) / 256, 256, 0, stream>>>(ewihr, Wh + OFF_EWIH1, 524288);
  rnnae_cvt_f16<<<(786432 + 255) / 256, 256, 0, stream>>>(ewhh,  Wh + OFF_EWHH,  786432);
  rnnae_cvt_f16<<<(786432 + 255) / 256, 256, 0, stream>>>(dwih,  Wh + OFF_DWIH,  786432);
  rnnae_cvt_f16<<<(786432 + 255) / 256, 256, 0, stream>>>(dwhh,  Wh + OFF_DWHH,  786432);
  rnnae_bias_sum<<<(3072 + 255) / 256, 256, 0, stream>>>(ebih, ebhh, dbih, dbhh, bse, bsd);
  rnnae_embed<<<(RNN_T * RNN_B + 255) / 256, 256, 0, stream>>>(x, emb, xs16);

  // persistent recurrent kernel: 1 workgroup, 32 wave32 waves, ~200KB LDS
  rnnae_persistent<<<1, 1024, LDS_BYTES, stream>>>(Wh, bse, bsd, xs16, proj,
                                                   (float*)d_out);
}
